// Bert_LSTMCRF_75771813036828
// MI455X (gfx1250) — matrix operand order repaired and checked by reading, not compiled, not run on hardware
//
#include <hip/hip_runtime.h>
#include <math.h>

// ---------------- problem constants ----------------
#define B_      256
#define T_      512
#define E_      768
#define HD_     50
#define G4_     200          // 4*Hd
#define G4P_    208          // padded to 13*16
#define KTAG    11
#define START_  9
#define STOP_   10
#define NEGV    (-1000.0f)
#define NT_DIR  13           // N tiles per direction
#define NT_TOT  26           // both directions
#define KSTEPS  (E_/32)      // 24 k-steps of 32 for bf16 WMMA
#define MROWS   (B_*T_)      // 131072

// ---------------- vector types (trivial, union-safe) ----------------
typedef __attribute__((ext_vector_type(16))) __bf16   v16bf;
typedef __attribute__((ext_vector_type(8)))  float    v8f;
typedef __attribute__((ext_vector_type(4))) unsigned  u32x4;
typedef __attribute__((ext_vector_type(4)))  float    f32x4;
typedef __attribute__((ext_vector_type(4))) unsigned  ui32x4_t;
typedef __attribute__((ext_vector_type(4)))  int      i32x4_t;
typedef __attribute__((ext_vector_type(8)))  int      i32x8_t;

union ABf { v16bf v; unsigned u[8]; unsigned short s[16]; };
union CDf { v8f v; float e[8]; };

__device__ __forceinline__ unsigned short f2bf(float f) {
  unsigned u = __float_as_uint(f);
  unsigned r = u + 0x7FFFu + ((u >> 16) & 1u);   // round-to-nearest-even
  return (unsigned short)(r >> 16);
}
__device__ __forceinline__ unsigned pkbf(float a, float b) {
  return (unsigned)f2bf(a) | ((unsigned)f2bf(b) << 16);
}
__device__ __forceinline__ float sigf(float x) { return 1.0f / (1.0f + __expf(-x)); }

#if defined(__has_builtin)
#if __has_builtin(__builtin_amdgcn_tensor_load_to_lds) && \
    __has_builtin(__builtin_amdgcn_s_wait_tensorcnt)
#define USE_TDM 1
#endif
#endif

// ---------------- K0: weight conversion to bf16 ----------------
__global__ void cvt_weights_kernel(const float* __restrict__ Wihf,
                                   const float* __restrict__ Wihb,
                                   const float* __restrict__ Whhf,
                                   const float* __restrict__ Whhb,
                                   unsigned short* __restrict__ WbIh,
                                   unsigned short* __restrict__ WbHh) {
  int idx = blockIdx.x * blockDim.x + threadIdx.x;
  int stride = gridDim.x * blockDim.x;
  const int nIh = 2 * G4P_ * E_;
  for (int i = idx; i < nIh; i += stride) {
    int d = i / (G4P_ * E_);
    int r = i - d * (G4P_ * E_);
    int n = r / E_, k = r - n * E_;
    float v = (n < G4_) ? (d ? Wihb : Wihf)[n * E_ + k] : 0.0f;
    WbIh[i] = f2bf(v);
  }
  const int nHh = 2 * G4P_ * 64;
  for (int i = idx; i < nHh; i += stride) {
    int d = i / (G4P_ * 64);
    int r = i - d * (G4P_ * 64);
    int n = r >> 6, k = r & 63;
    float v = (n < G4_ && k < HD_) ? (d ? Whhb : Whhf)[n * HD_ + k] : 0.0f;
    WbHh[i] = f2bf(v);
  }
}

// ---------------- K1: big input-gate GEMM via WMMA bf16 ----------------
// One block = one 16-row M tile; 4 waves cover all 26 N tiles.
// Scalar (readfirstlane) wave id -> uniform scalar branches, no EXEC churn.
// Two-phase k-step: issue all 14 B-fragment b128 loads, then 7 chained WMMAs.
__global__ __launch_bounds__(128) void gates_gemm_kernel(
    const float* __restrict__ X,                 // [MROWS, E]
    const unsigned short* __restrict__ WbIh,     // [2][208*768] bf16 padded
    const float* __restrict__ bf_bias,
    const float* __restrict__ bb_bias,
    float* __restrict__ G) {                     // [2][MROWS*208] padded stride
  const int tid   = threadIdx.x;
  const int wid   = __builtin_amdgcn_readfirstlane(threadIdx.x) >> 5;  // SGPR
  const int lane  = tid & 31;
  const int laneM = lane & 15;
  const int hi    = lane >> 4;
  const long rowbase = (long)blockIdx.x * 16;
  const float* xrow = X + (rowbase + laneM) * E_;

  CDf acc[7];
#pragma unroll
  for (int i = 0; i < 7; ++i)
#pragma unroll
    for (int r = 0; r < 8; ++r) acc[i].e[r] = 0.0f;

  for (int ks = 0; ks < KSTEPS; ++ks) {
    const int kb = ks * 32;
    // A fragment: 16 bf16 per lane, converted from f32 in-register.
    ABf a;
    {
      f32x4 f0 = *(const f32x4*)(xrow + kb + hi * 8);
      f32x4 f1 = *(const f32x4*)(xrow + kb + hi * 8 + 4);
      f32x4 f2 = *(const f32x4*)(xrow + kb + 16 + hi * 8);
      f32x4 f3 = *(const f32x4*)(xrow + kb + 16 + hi * 8 + 4);
      a.u[0] = pkbf(f0.x, f0.y); a.u[1] = pkbf(f0.z, f0.w);
      a.u[2] = pkbf(f1.x, f1.y); a.u[3] = pkbf(f1.z, f1.w);
      a.u[4] = pkbf(f2.x, f2.y); a.u[5] = pkbf(f2.z, f2.w);
      a.u[6] = pkbf(f3.x, f3.y); a.u[7] = pkbf(f3.z, f3.w);
    }
    // phase 1: all B fragments for this k-step
    ABf b[7];
#pragma unroll
    for (int i = 0; i < 7; ++i) {
      const int nt = wid + 4 * i;
      if (nt >= NT_TOT) continue;           // scalar branch
      const int dir = nt / NT_DIR;
      const int n   = (nt % NT_DIR) * 16 + laneM;   // < 208, pad rows are zero
      const unsigned short* wrow = WbIh + ((long)dir * G4P_ + n) * E_;
      u32x4 b0 = *(const u32x4*)(wrow + kb + hi * 8);
      u32x4 b1 = *(const u32x4*)(wrow + kb + 16 + hi * 8);
      b[i].u[0] = b0.x; b[i].u[1] = b0.y; b[i].u[2] = b0.z; b[i].u[3] = b0.w;
      b[i].u[4] = b1.x; b[i].u[5] = b1.y; b[i].u[6] = b1.z; b[i].u[7] = b1.w;
    }
    // phase 2: chained WMMAs (XDL overlaps next iteration's loads)
#pragma unroll
    for (int i = 0; i < 7; ++i) {
      const int nt = wid + 4 * i;
      if (nt >= NT_TOT) continue;
      acc[i].v = __builtin_amdgcn_wmma_f32_16x16x32_bf16(
          false, a.v, false, b[i].v, (short)0, acc[i].v, false, false);
    }
  }
  // store with bias fold (padded stride; only n<200 written)
#pragma unroll
  for (int i = 0; i < 7; ++i) {
    const int nt = wid + 4 * i;
    if (nt >= NT_TOT) continue;
    const int dir = nt / NT_DIR;
    const int n   = (nt % NT_DIR) * 16 + laneM;
    if (n >= G4_) continue;
    const float bias = (dir ? bb_bias : bf_bias)[n];
    float* Gd = G + (long)dir * MROWS * G4P_;
#pragma unroll
    for (int r = 0; r < 8; ++r) {
      long row = rowbase + r + hi * 8;
      Gd[row * G4P_ + n] = acc[i].e[r] + bias;
    }
  }
}

// ---------------- K2: LSTM recurrence (WMMA on h @ W_hh^T) ----------------
// 32 blocks: blockIdx = dir*16 + batch-group (16 rows each). 4 waves/block.
__global__ __launch_bounds__(128) void lstm_rec_kernel(
    const unsigned short* __restrict__ WbHh,  // [2][208*64] bf16 padded
    const float* __restrict__ G,              // [2][MROWS*208] padded
    float* __restrict__ hs) {                 // [MROWS*100]
  __shared__ unsigned short Wsm[G4P_ * 64];   // 26.0 KB
  __shared__ unsigned short hA[16 * 64];      // h (bf16), k-padded to 64
  __shared__ float gsm[16 * G4P_];            // gate pre-activations
  const int tid   = threadIdx.x;
  const int wid   = __builtin_amdgcn_readfirstlane(threadIdx.x) >> 5;  // SGPR
  const int lane  = tid & 31;
  const int laneM = lane & 15;
  const int hi    = lane >> 4;
  const int dir     = blockIdx.x >> 4;
  const int rowbase = (blockIdx.x & 15) * 16;
  const float* Gd = G + (long)dir * MROWS * G4P_;
  const unsigned short* Wd = WbHh + dir * G4P_ * 64;

#if defined(USE_TDM)
  // Tensor Data Mover: async DMA of the W_hh tile into LDS (TENSORcnt).
  // Manual copy below remains the correctness backstop.
  if (tid == 0) {
    unsigned ldsAddr = (unsigned)(unsigned long long)(&Wsm[0]);
    unsigned long long ga = (unsigned long long)Wd;
    ui32x4_t g0;
    g0.x = 0x1u;                                   // count=1 user descriptor
    g0.y = ldsAddr;                                // lds_addr
    g0.z = (unsigned)(ga & 0xFFFFFFFFu);           // global_addr[31:0]
    g0.w = (unsigned)((ga >> 32) & 0x01FFFFFFu) | 0x80000000u;  // msbs + type=2
    i32x8_t g1;
    const int elems = G4P_ * 64;                   // 13312 bf16 elements
    g1[0] = (1 << 16);                             // data_size=1 (2B)
    g1[1] = (elems & 0xFFFF) << 16;                // tensor_dim0 lo
    g1[2] = ((elems >> 16) & 0xFFFF) | (1 << 16);  // dim0 hi | tensor_dim1=1
    g1[3] = (elems & 0xFFFF) << 16;                // tile_dim0 = elems
    g1[4] = 1;                                     // tile_dim1=1
    g1[5] = elems;                                 // tensor_dim0_stride lo
    g1[6] = 0;
    g1[7] = 0;
    i32x4_t z4 = {0, 0, 0, 0};
#if __clang_major__ >= 23
    i32x8_t z8 = {0, 0, 0, 0, 0, 0, 0, 0};
    __builtin_amdgcn_tensor_load_to_lds(g0, g1, z4, z4, z8, 0);
#else
    __builtin_amdgcn_tensor_load_to_lds(g0, g1, z4, z4, 0);
#endif
  }
  __builtin_amdgcn_s_wait_tensorcnt(0);
#endif
  for (int i = tid; i < G4P_ * 64 / 2; i += 128)
    ((unsigned*)Wsm)[i] = ((const unsigned*)Wd)[i];
  for (int i = tid; i < 16 * 64; i += 128) hA[i] = 0;
  float cst[7];
#pragma unroll
  for (int i = 0; i < 7; ++i) cst[i] = 0.0f;
  __syncthreads();

  for (int t = 0; t < T_; ++t) {
    const int tact = dir ? (T_ - 1 - t) : t;
    if (t + 1 < T_) {   // L2 prefetch of next step's gate rows
      int tn = dir ? (T_ - 2 - t) : (t + 1);
      if (lane < 16)
        __builtin_prefetch(&Gd[((long)(rowbase + laneM) * T_ + tn) * G4P_], 0, 1);
    }
    // A fragments from hA: two k-halves of the padded K=64
    ABf a0, a1;
    {
      u32x4 q;
      q = *(const u32x4*)&hA[laneM * 64 + hi * 8];
      a0.u[0] = q.x; a0.u[1] = q.y; a0.u[2] = q.z; a0.u[3] = q.w;
      q = *(const u32x4*)&hA[laneM * 64 + 16 + hi * 8];
      a0.u[4] = q.x; a0.u[5] = q.y; a0.u[6] = q.z; a0.u[7] = q.w;
      q = *(const u32x4*)&hA[laneM * 64 + 32 + hi * 8];
      a1.u[0] = q.x; a1.u[1] = q.y; a1.u[2] = q.z; a1.u[3] = q.w;
      q = *(const u32x4*)&hA[laneM * 64 + 48 + hi * 8];
      a1.u[4] = q.x; a1.u[5] = q.y; a1.u[6] = q.z; a1.u[7] = q.w;
    }
    // phase 1: C-init loads (padded stride, no per-lane guard) + B fragments
    CDf acc[4];
    ABf b0[4], b1[4];
#pragma unroll
    for (int i = 0; i < 4; ++i) {
      const int nt = wid + 4 * i;
      if (nt >= NT_DIR) continue;           // scalar branch
      const int n = nt * 16 + laneM;
#pragma unroll
      for (int r = 0; r < 8; ++r) {
        long b = rowbase + r + hi * 8;
        acc[i].e[r] = Gd[(b * T_ + tact) * G4P_ + n];
      }
      u32x4 qb;
      qb = *(const u32x4*)&Wsm[n * 64 + hi * 8];
      b0[i].u[0] = qb.x; b0[i].u[1] = qb.y; b0[i].u[2] = qb.z; b0[i].u[3] = qb.w;
      qb = *(const u32x4*)&Wsm[n * 64 + 16 + hi * 8];
      b0[i].u[4] = qb.x; b0[i].u[5] = qb.y; b0[i].u[6] = qb.z; b0[i].u[7] = qb.w;
      qb = *(const u32x4*)&Wsm[n * 64 + 32 + hi * 8];
      b1[i].u[0] = qb.x; b1[i].u[1] = qb.y; b1[i].u[2] = qb.z; b1[i].u[3] = qb.w;
      qb = *(const u32x4*)&Wsm[n * 64 + 48 + hi * 8];
      b1[i].u[4] = qb.x; b1[i].u[5] = qb.y; b1[i].u[6] = qb.z; b1[i].u[7] = qb.w;
    }
    // phase 2: WMMAs + gsm scatter
#pragma unroll
    for (int i = 0; i < 4; ++i) {
      const int nt = wid + 4 * i;
      if (nt >= NT_DIR) continue;
      acc[i].v = __builtin_amdgcn_wmma_f32_16x16x32_bf16(
          false, a0.v, false, b0[i].v, (short)0, acc[i].v, false, false);
      acc[i].v = __builtin_amdgcn_wmma_f32_16x16x32_bf16(
          false, a1.v, false, b1[i].v, (short)0, acc[i].v, false, false);
      const int n = nt * 16 + laneM;
#pragma unroll
      for (int r = 0; r < 8; ++r)
        gsm[(r + hi * 8) * G4P_ + n] = acc[i].e[r];
    }
    __syncthreads();
    // elementwise cell update (torch gate order i,f,g,o)
#pragma unroll
    for (int i = 0; i < 7; ++i) {
      const int e = tid + i * 128;
      if (e < 16 * HD_) {
        const int m = e / HD_, j = e - m * HD_;
        float ig = gsm[m * G4P_ + j];
        float fg = gsm[m * G4P_ + HD_ + j];
        float gg = gsm[m * G4P_ + 2 * HD_ + j];
        float og = gsm[m * G4P_ + 3 * HD_ + j];
        float c = sigf(fg) * cst[i] + sigf(ig) * tanhf(gg);
        float h = sigf(og) * tanhf(c);
        cst[i] = c;
        hA[m * 64 + j] = f2bf(h);
        long b = rowbase + m;
        hs[(b * T_ + tact) * 100 + dir * HD_ + j] = h;
      }
    }
    __syncthreads();
  }
}

// ---------------- K3a: output projection ----------------
__global__ __launch_bounds__(256) void proj_logits_kernel(
    const float* __restrict__ hs, const float* __restrict__ Wout,
    const float* __restrict__ bout, float* __restrict__ logits) {
  __shared__ float sW[KTAG * 100];
  __shared__ float sB[KTAG];
  for (int i = threadIdx.x; i < KTAG * 100; i += 256) sW[i] = Wout[i];
  if (threadIdx.x < KTAG) sB[threadIdx.x] = bout[threadIdx.x];
  __syncthreads();
  long row = (long)blockIdx.x * 256 + threadIdx.x;
  if (row >= MROWS) return;
  float acc[KTAG];
#pragma unroll
  for (int k = 0; k < KTAG; ++k) acc[k] = sB[k];
  const float* hv = hs + row * 100;
  for (int j = 0; j < 100; ++j) {
    float h = hv[j];
#pragma unroll
    for (int k = 0; k < KTAG; ++k) acc[k] += h * sW[k * 100 + j];
  }
#pragma unroll
  for (int k = 0; k < KTAG; ++k) logits[row * KTAG + k] = acc[k];
}

// ---------------- K3b: CRF forward (logsumexp) + Viterbi backtrace ----------------
__global__ __launch_bounds__(32) void crf_kernel(
    const float* __restrict__ logits, const int* __restrict__ mask,
    const float* __restrict__ trans, int* __restrict__ bps,
    float* __restrict__ scores, int* __restrict__ paths) {
  const int b = blockIdx.x;
  const int lane = threadIdx.x;
  __shared__ float sA[KTAG];
  __shared__ float sT[KTAG * KTAG];
  for (int i = lane; i < KTAG * KTAG; i += 32) sT[i] = trans[i];
  if (lane < KTAG) sA[lane] = (lane == START_) ? 0.0f : NEGV;
  __syncthreads();
  for (int t = 0; t < T_; ++t) {
    float al[KTAG];
#pragma unroll
    for (int p = 0; p < KTAG; ++p) al[p] = sA[p];
    float na = 0.0f;
    int bp = lane;
    if (lane < KTAG) {
      const int m = mask[b * T_ + t];
      float frame = logits[((long)b * T_ + t) * KTAG + lane];
      float mx = -INFINITY;
      int arg = 0;
#pragma unroll
      for (int p = 0; p < KTAG; ++p) {
        float s = al[p] + sT[p * KTAG + lane];
        if (s > mx) { mx = s; arg = p; }
      }
      float sum = 0.0f;
#pragma unroll
      for (int p = 0; p < KTAG; ++p)
        sum += __expf(al[p] + sT[p * KTAG + lane] - mx);
      na = frame + mx + __logf(sum);
      if (m) { bp = arg; } else { na = al[lane]; bp = lane; }
    }
    __syncthreads();
    if (lane < KTAG) {
      sA[lane] = na;
      bps[((long)t * B_ + b) * KTAG + lane] = bp;
    }
    __syncthreads();
  }
  if (lane == 0) {
    float mx = -INFINITY;
    int best = 0;
    float fin[KTAG];
    for (int k = 0; k < KTAG; ++k) {
      fin[k] = sA[k] + sT[k * KTAG + STOP_];
      if (fin[k] > mx) { mx = fin[k]; best = k; }
    }
    float sum = 0.0f;
    for (int k = 0; k < KTAG; ++k) sum += __expf(fin[k] - mx);
    scores[b] = mx + __logf(sum);
    int cur = best;
    paths[b * T_ + (T_ - 1)] = best;
    for (int t = T_ - 1; t >= 1; --t) {
      int prev = bps[((long)t * B_ + b) * KTAG + cur];
      paths[b * T_ + (t - 1)] = prev;
      cur = prev;
    }
  }
}

// ---------------- host launcher ----------------
extern "C" void kernel_launch(void* const* d_in, const int* in_sizes, int n_in,
                              void* d_out, int out_size, void* d_ws, size_t ws_size,
                              hipStream_t stream) {
  (void)in_sizes; (void)n_in; (void)out_size; (void)ws_size;
  const float* X     = (const float*)d_in[0];
  const int*   msk   = (const int*)d_in[1];
  const float* Wihf  = (const float*)d_in[2];
  const float* Whhf  = (const float*)d_in[3];
  const float* bfb   = (const float*)d_in[4];
  const float* Wihb  = (const float*)d_in[5];
  const float* Whhb  = (const float*)d_in[6];
  const float* bbb   = (const float*)d_in[7];
  const float* Wout  = (const float*)d_in[8];
  const float* bout  = (const float*)d_in[9];
  const float* trans = (const float*)d_in[10];

  char* ws = (char*)d_ws;
  size_t off = 0;
  auto carve = [&](size_t bytes) {
    char* p = ws + off;
    off = (off + bytes + 255) & ~(size_t)255;
    return p;
  };
  unsigned short* WbIh = (unsigned short*)carve((size_t)2 * G4P_ * E_ * 2);
  unsigned short* WbHh = (unsigned short*)carve((size_t)2 * G4P_ * 64 * 2);
  float* G      = (float*)carve((size_t)2 * MROWS * G4P_ * 4);
  float* hsbuf  = (float*)carve((size_t)MROWS * 100 * 4);
  float* logits = (float*)carve((size_t)MROWS * KTAG * 4);
  int*   bps    = (int*)carve((size_t)T_ * B_ * KTAG * 4);

  float* scores = (float*)d_out;          // [B]
  int*   paths  = (int*)d_out + B_;       // [B,T] as raw int32 bits

  cvt_weights_kernel<<<512, 256, 0, stream>>>(Wihf, Wihb, Whhf, Whhb, WbIh, WbHh);
  gates_gemm_kernel<<<MROWS / 16, 128, 0, stream>>>(X, WbIh, bfb, bbb, G);
  lstm_rec_kernel<<<32, 128, 0, stream>>>(WbHh, G, hsbuf);
  proj_logits_kernel<<<(MROWS + 255) / 256, 256, 0, stream>>>(hsbuf, Wout, bout, logits);
  crf_kernel<<<B_, 32, 0, stream>>>(logits, msk, trans, bps, scores, paths);
}
